// Sentiment_Attention_40475771798179
// MI455X (gfx1250) — compile-verified, matrix operand order checked by reading, not run on hardware
//
#include <hip/hip_runtime.h>
#include <math.h>

// ---------------- problem constants ----------------
constexpr int Bn = 32;     // batch
constexpr int Sn = 2048;   // sequence
constexpr int En = 1024;   // encoder dim
constexpr int An = 512;    // attention dim
constexpr int NS = 8;      // S-splits (flash-decoding style)
constexpr int SC = Sn / NS;    // 256 s-values per split
constexpr int TS = 64;         // s-tile rows (4 WMMA M-tiles)
constexpr int MT = TS / 16;    // 4
constexpr int KK = En / 32;    // 32 k-steps of 32 (bf16 WMMA K)
constexpr int NTILES = 2 * (An / 16); // 64 (head,ntile) column tiles

typedef __attribute__((ext_vector_type(16))) __bf16 v16bf;
typedef __attribute__((ext_vector_type(8)))  float  v8f;

static __device__ inline v8f wmma_bf16(v16bf a, v16bf b, v8f c) {
  return __builtin_amdgcn_wmma_f32_16x16x32_bf16(false, a, false, b, (short)0, c,
                                                 false, false);
}

static __device__ inline unsigned short f2bf(float f) {
  unsigned u = __float_as_uint(f);
  u += 0x7fffu + ((u >> 16) & 1u);   // round-to-nearest-even
  return (unsigned short)(u >> 16);
}

static __device__ inline float fast_tanh(float x) {
  float ax = fabsf(x);
  float e  = __expf(2.f * ax);                 // inf-safe: e=inf -> t=1
  float t  = 1.f - 2.f / (e + 1.f);
  return copysignf(t, x);
}

// bf16 A/B fragment index packing (ISA 7.12.2, 16-bit 16x32):
// lane l holds K = (j/8)*16 + (l/16)*8 + (j%8), row/col = l%16.

// ---------------- kernel 0: prepack W into B-fragment order (bf16) ---------
__global__ __launch_bounds__(256) void prepack_w(const float* __restrict__ W1,
                                                 const float* __restrict__ W2,
                                                 unsigned short* __restrict__ wpk) {
  int idx = blockIdx.x * 256 + threadIdx.x;
  const int total = 2 * An * En;
  if (idx >= total) return;
  int head = idx / (An * En);
  int r    = idx - head * (An * En);
  int a = r / En, e = r - a * En;
  float v = (head ? W2 : W1)[r];
  int ntile = a >> 4, n = a & 15;
  int kk = e >> 5, ke = e & 31;
  int j    = ((ke >> 4) << 3) | (ke & 7);
  int lane = (((ke >> 3) & 1) << 4) | n;
  int g = head * (An / 16) + ntile;            // 0..63
  size_t off = ((((size_t)g * KK + kk) << 5) | lane) * 16 + j;
  wpk[off] = f2bf(v);
}

// ---------------- kernel 1: fused proj-GEMM + online softmax + context -----
__global__ __launch_bounds__(256) void attn_main(
    const float* __restrict__ enc, const float* __restrict__ arg1,
    const float* __restrict__ arg2, const unsigned char* __restrict__ mask,
    const float* __restrict__ b1, const float* __restrict__ b2,
    const unsigned short* __restrict__ wpk, float* __restrict__ pm,
    float* __restrict__ pl, float* __restrict__ pctx) {
  __shared__ __align__(32) unsigned short encA[TS * En];  // 128KB, frag order
  __shared__ float sc[2][TS];
  __shared__ float psc[2][TS];
  __shared__ float msh[2], lsh[2], fsh[2];

  const int tid  = threadIdx.x;
  const int wave = tid >> 5;
  const int lane = tid & 31;
  const int b     = blockIdx.y;
  const int split = blockIdx.x;

  if (tid < 2) { msh[tid] = -INFINITY; lsh[tid] = 0.f; }
  if (tid < 2 * TS) ((float*)sc)[tid] = 0.f;

  // persistent per-thread context accumulators: e-range [tid*4, tid*4+4)
  float c1x = 0.f, c1y = 0.f, c1z = 0.f, c1w = 0.f;
  float c2x = 0.f, c2y = 0.f, c2z = 0.f, c2w = 0.f;

  for (int st = 0; st < SC / TS; ++st) {
    const int s_base = split * SC + st * TS;
    const float* src = enc + ((size_t)b * Sn + s_base) * En;

    // ---- stage enc tile -> LDS (bf16, A-fragment scatter), coalesced f4 ----
    for (int i4 = tid; i4 < TS * En / 4; i4 += 256) {
      int sl = i4 >> 8;                 // /(En/4)
      int e0 = (i4 & (En / 4 - 1)) << 2;
      const float4 v = *(const float4*)(src + (size_t)sl * En + e0);
      int kk = e0 >> 5, ke = e0 & 31;
      int j0   = ((ke >> 4) << 3) | (ke & 7);
      int lid  = (((ke >> 3) & 1) << 4) | (sl & 15);
      int mt   = sl >> 4;
      unsigned off = ((((mt * KK + kk) << 5) | lid) << 4) + j0;
      encA[off + 0] = f2bf(v.x); encA[off + 1] = f2bf(v.y);
      encA[off + 2] = f2bf(v.z); encA[off + 3] = f2bf(v.w);
    }
    __syncthreads();

    // ---- GEMM: wave owns 8 (head,ntile) column tiles, 2 at a time ----
    for (int ntc = 0; ntc < 4; ++ntc) {
      const int g0 = (wave << 3) + ntc * 2, g1 = g0 + 1;
      v8f acc0[MT], acc1[MT];
#pragma unroll
      for (int mt = 0; mt < MT; ++mt) {
        acc0[mt] = (v8f){0.f, 0.f, 0.f, 0.f, 0.f, 0.f, 0.f, 0.f};
        acc1[mt] = (v8f){0.f, 0.f, 0.f, 0.f, 0.f, 0.f, 0.f, 0.f};
      }
      const unsigned short* bp0 = wpk + (((size_t)g0 * KK) << 9) + lane * 16;
      const unsigned short* bp1 = wpk + (((size_t)g1 * KK) << 9) + lane * 16;
      for (int kk = 0; kk < KK; ++kk) {
        v16bf bf0 = *(const v16bf*)(bp0 + (kk << 9));
        v16bf bf1 = *(const v16bf*)(bp1 + (kk << 9));
#pragma unroll
        for (int mt = 0; mt < MT; ++mt) {
          v16bf af = *(const v16bf*)(&encA[(((mt * KK + kk) << 5) | lane) << 4]);
          acc0[mt] = wmma_bf16(af, bf0, acc0[mt]);
          acc1[mt] = wmma_bf16(af, bf1, acc1[mt]);
        }
      }
      // ---- epilogue: bias + tanh, *arg, row-reduce -> scores ----
#pragma unroll
      for (int q = 0; q < 2; ++q) {
        const int g = q ? g1 : g0;
        const int head = g >> 5, ntile = g & 31;
        const int a = (ntile << 4) + (lane & 15);
        const float argv = (head ? arg2 : arg1)[b * An + a];
        const float bias = (head ? b2 : b1)[a];
        const int rowoff = (lane >> 4) << 3;
#pragma unroll
        for (int mt = 0; mt < MT; ++mt) {
          v8f acc = q ? acc1[mt] : acc0[mt];
#pragma unroll
          for (int r = 0; r < 8; ++r) {
            float v = fast_tanh(acc[r] + bias) * argv;
            v += __shfl_xor(v, 1); v += __shfl_xor(v, 2);
            v += __shfl_xor(v, 4); v += __shfl_xor(v, 8);
            if ((lane & 15) == 0)
              atomicAdd(&sc[head][mt * 16 + r + rowoff], v);
          }
        }
      }
    }
    __syncthreads();  // scores complete

    // ---- online softmax update (wave h handles head h) ----
    if (wave < 2) {
      const int h = wave;
      float v0 = sc[h][lane], v1 = sc[h][lane + 32];
      if (mask[(size_t)b * Sn + s_base + lane])      v0 = -INFINITY;
      if (mask[(size_t)b * Sn + s_base + lane + 32]) v1 = -INFINITY;
      float tmax = fmaxf(v0, v1);
      tmax = fmaxf(tmax, __shfl_xor(tmax, 1));
      tmax = fmaxf(tmax, __shfl_xor(tmax, 2));
      tmax = fmaxf(tmax, __shfl_xor(tmax, 4));
      tmax = fmaxf(tmax, __shfl_xor(tmax, 8));
      tmax = fmaxf(tmax, __shfl_xor(tmax, 16));
      const float mo = msh[h], lo = lsh[h];
      const float mn = fmaxf(mo, tmax);
      const float fac = (lo > 0.f) ? __expf(mo - mn) : 0.f;
      float p0 = (v0 == -INFINITY) ? 0.f : __expf(v0 - mn);
      float p1 = (v1 == -INFINITY) ? 0.f : __expf(v1 - mn);
      psc[h][lane] = p0; psc[h][lane + 32] = p1;
      float ps = p0 + p1;
      ps += __shfl_xor(ps, 1); ps += __shfl_xor(ps, 2);
      ps += __shfl_xor(ps, 4); ps += __shfl_xor(ps, 8);
      ps += __shfl_xor(ps, 16);
      if (lane == 0) { msh[h] = mn; lsh[h] = lo * fac + ps; fsh[h] = fac; }
    }
    __syncthreads();

    // ---- context accumulate (coalesced re-read of the L2-hot tile) ----
    {
      const float f0 = fsh[0], f1 = fsh[1];
      c1x *= f0; c1y *= f0; c1z *= f0; c1w *= f0;
      c2x *= f1; c2y *= f1; c2z *= f1; c2w *= f1;
      const float4* ep = (const float4*)src + tid;  // e-range tid*4..+3
#pragma unroll 4
      for (int s = 0; s < TS; ++s) {
        const float p0 = psc[0][s], p1 = psc[1][s];
        const float4 v = ep[s * (En / 4)];
        c1x += p0 * v.x; c1y += p0 * v.y; c1z += p0 * v.z; c1w += p0 * v.w;
        c2x += p1 * v.x; c2y += p1 * v.y; c2z += p1 * v.z; c2w += p1 * v.w;
      }
    }
    __syncthreads();            // done with psc/encA for this tile
    if (tid < 2 * TS) ((float*)sc)[tid] = 0.f;  // reset before next tile
  }

  // ---- write partials ----
  const size_t pidx = ((size_t)b * NS + split) * 2;
  if (tid == 0) { pm[pidx] = msh[0]; pl[pidx] = lsh[0];
                  pm[pidx + 1] = msh[1]; pl[pidx + 1] = lsh[1]; }
  float* d0 = pctx + pidx * En + tid * 4;
  d0[0] = c1x; d0[1] = c1y; d0[2] = c1z; d0[3] = c1w;
  float* d1 = pctx + (pidx + 1) * En + tid * 4;
  d1[0] = c2x; d1[1] = c2y; d1[2] = c2z; d1[3] = c2w;
}

// ---------------- kernel 2: combine split partials ----------------
__global__ __launch_bounds__(256) void combine(const float* __restrict__ pm,
                                               const float* __restrict__ pl,
                                               const float* __restrict__ pctx,
                                               float* __restrict__ out) {
  const int b = blockIdx.x, tid = threadIdx.x;
#pragma unroll
  for (int h = 0; h < 2; ++h) {
    float M = -INFINITY;
    for (int i = 0; i < NS; ++i)
      M = fmaxf(M, pm[((size_t)b * NS + i) * 2 + h]);
    float L = 0.f, w[NS];
    for (int i = 0; i < NS; ++i) {
      const size_t k = ((size_t)b * NS + i) * 2 + h;
      float wi = (pm[k] == -INFINITY) ? 0.f : __expf(pm[k] - M);
      w[i] = wi;
      L += wi * pl[k];
    }
    const float inv = (L > 0.f) ? 1.f / L : 0.f;
    const int e0 = tid * 4;
    float ax = 0.f, ay = 0.f, az = 0.f, aw = 0.f;
    for (int i = 0; i < NS; ++i) {
      const float4 v =
          *(const float4*)(pctx + (((size_t)b * NS + i) * 2 + h) * En + e0);
      ax += w[i] * v.x; ay += w[i] * v.y; az += w[i] * v.z; aw += w[i] * v.w;
    }
    float* o = out + (size_t)b * (2 * En) + h * En + e0;
    o[0] = ax * inv; o[1] = ay * inv; o[2] = az * inv; o[3] = aw * inv;
  }
}

// ---------------- launcher ----------------
extern "C" void kernel_launch(void* const* d_in, const int* in_sizes, int n_in,
                              void* d_out, int out_size, void* d_ws,
                              size_t ws_size, hipStream_t stream) {
  const float* arg1 = (const float*)d_in[0];
  const float* arg2 = (const float*)d_in[1];
  const float* enc  = (const float*)d_in[2];
  const unsigned char* mask = (const unsigned char*)d_in[3];
  const float* W1 = (const float*)d_in[4];
  const float* b1 = (const float*)d_in[5];
  const float* W2 = (const float*)d_in[6];
  const float* b2 = (const float*)d_in[7];
  float* out = (float*)d_out;

  const size_t wpk_bytes = (size_t)2 * An * En * sizeof(unsigned short);
  unsigned short* wpk = (unsigned short*)d_ws;
  float* pm = (float*)((char*)d_ws + wpk_bytes);
  float* pl = pm + (size_t)Bn * NS * 2;
  float* pctx = pl + (size_t)Bn * NS * 2;

  prepack_w<<<(2 * An * En + 255) / 256, 256, 0, stream>>>(W1, W2, wpk);
  attn_main<<<dim3(NS, Bn), 256, 0, stream>>>(enc, arg1, arg2, mask, b1, b2,
                                              wpk, pm, pl, pctx);
  combine<<<Bn, 256, 0, stream>>>(pm, pl, pctx, out);
}
